// IterativeRecoModel_70660801953987
// MI455X (gfx1250) — compile-verified, matrix operand order checked by reading, not run on hardware
//
#include <hip/hip_runtime.h>
#include <hip/hip_bf16.h>

// CDNA5 / gfx1250: wave32, v_wmma_f32_16x16x32_bf16
typedef __attribute__((ext_vector_type(16))) __bf16 v16bf;
typedef __attribute__((ext_vector_type(8)))  float  v8f;

#define IMG_H     512
#define IMG_W     512
#define N_VIEWS   180
#define N_DET     512
#define NSAMP     512
#define DET_SPACING 2.0f
#define HALF_W    255.5f          // (W-1)/2
#define HALF_EXT  256.5f          // bilinear support: px in [-1, 512]
#define PAD       2
#define PW        (IMG_W + 2*PAD) // 516, zero ring of >=2 on all sides
#define PH        (IMG_H + 2*PAD)

// ---------------------------------------------------------------------------
// Kernel 1a (staged): build zero-padded updated image in ws + relu output.
// Grid: (ceil(PW/256), PH), block 256.
// ---------------------------------------------------------------------------
__global__ void prep_padded(const float* __restrict__ x,
                            const float* __restrict__ reco,
                            float* __restrict__ pimg,
                            float* __restrict__ relu) {
    int c = blockIdx.x * blockDim.x + threadIdx.x;
    int r = blockIdx.y;
    if (c >= PW) return;
    int ri = r - PAD, ci = c - PAD;
    float v = 0.0f;
    if (((unsigned)ri < (unsigned)IMG_H) & ((unsigned)ci < (unsigned)IMG_W)) {
        int idx = (ri << 9) + ci;
        v = x[idx] + reco[idx];
        relu[idx] = fmaxf(v, 0.0f);
    }
    pimg[r * PW + c] = v;
}

// Kernel 1b (fallback, no usable scratch): relu only.
__global__ void prep_simple(const float4* __restrict__ x,
                            const float4* __restrict__ reco,
                            float4* __restrict__ relu) {
    int i = blockIdx.x * blockDim.x + threadIdx.x;
    float4 a = x[i], b = reco[i];
    relu[i] = make_float4(fmaxf(a.x + b.x, 0.f), fmaxf(a.y + b.y, 0.f),
                          fmaxf(a.z + b.z, 0.f), fmaxf(a.w + b.w, 0.f));
}

// ---------------------------------------------------------------------------
// Gathers.
// Padded path: clamp-only, 4 unconditional loads; zero ring => exact
// zero-outside semantics for arbitrarily far-outside positions.
// ---------------------------------------------------------------------------
__device__ __forceinline__ float bilin_pad(const float* __restrict__ img,
                                           float pxp, float pyp) {
    float fx0 = floorf(pxp), fy0 = floorf(pyp);
    float fx = pxp - fx0, fy = pyp - fy0;
    int ix0 = (int)fx0, iy0 = (int)fy0;
    ix0 = min(max(ix0, 0), PW - 2);
    iy0 = min(max(iy0, 0), PH - 2);
    const float* p = img + iy0 * PW + ix0;
    float v00 = p[0], v10 = p[1], v01 = p[PW], v11 = p[PW + 1];
    float top = fmaf(fx, v10 - v00, v00);
    float bot = fmaf(fx, v11 - v01, v01);
    return fmaf(fy, bot - top, top);
}

// Fallback path: checked gather over x+reco.
__device__ __forceinline__ float fetch2(const float* __restrict__ a,
                                        const float* __restrict__ b,
                                        int iy, int ix) {
    bool inb = ((unsigned)ix < (unsigned)IMG_W) & ((unsigned)iy < (unsigned)IMG_H);
    int cx = min(max(ix, 0), IMG_W - 1);
    int cy = min(max(iy, 0), IMG_H - 1);
    int idx = (cy << 9) + cx;
    float t = a[idx] + b[idx];
    return inb ? t : 0.0f;
}
__device__ __forceinline__ float bilin_chk(const float* __restrict__ a,
                                           const float* __restrict__ b,
                                           float px, float py) {
    float fx0 = floorf(px), fy0 = floorf(py);
    float fx = px - fx0, fy = py - fy0;
    int ix0 = (int)fx0, iy0 = (int)fy0;
    float v00 = fetch2(a, b, iy0,     ix0);
    float v10 = fetch2(a, b, iy0,     ix0 + 1);
    float v01 = fetch2(a, b, iy0 + 1, ix0);
    float v11 = fetch2(a, b, iy0 + 1, ix0 + 1);
    float top = fmaf(fx, v10 - v00, v00);
    float bot = fmaf(fx, v11 - v01, v01);
    return fmaf(fy, bot - top, top);
}

// ---------------------------------------------------------------------------
// Kernel 2: fan-beam projector. One wave per 16-detector tile of one view.
//
// WMMA reduction with A = ones(16x32), B = samples(32x16):
//   D[m,n] = sum_k B[k,n]  -> per-ray sums land one-per-COLUMN.
// B layout (ISA 7.12.5): lanes 0-15 supply K=0-7,16-23 of column n=lane,
// lanes 16-31 supply K=8-15,24-31 of column n=lane-16; element e <-> K=8h+e (e<8), 8+8h+e (e>=8).
// Lane (n, half) computes 8 distinct samples s = sb + half*8 + j of ray n and
// emits hi(bf16) in elements 0-7 and lo = s - hi in elements 8-15, so each
// f32 sample carries ~16 mantissa bits and no sample is computed twice.
// C/D layout maps lane n to column n => lane n's ray sum is in c[0].
// ---------------------------------------------------------------------------
template <bool FUSED>
__global__ void proj_kernel(const float* __restrict__ imgA,
                            const float* __restrict__ imgB,
                            const float* __restrict__ src,
                            const float* __restrict__ detc,
                            const float* __restrict__ duv,
                            float* __restrict__ sino) {
    const int wave = threadIdx.x >> 5;
    const int l    = threadIdx.x & 31;
    const int v    = blockIdx.x >> 2;                  // view
    const int tile = ((blockIdx.x & 3) << 3) + wave;   // 0..31
    const int d0   = tile << 4;
    const int m    = l & 15;                           // this lane's ray/column
    const int det  = d0 + m;

    // Ray geometry (lanes m and m+16 share ray m).
    const float sx = src[2 * v],  sy = src[2 * v + 1];
    const float u  = ((float)det - HALF_W) * DET_SPACING;
    const float ex = fmaf(u, duv[2 * v],     detc[2 * v]);
    const float ey = fmaf(u, duv[2 * v + 1], detc[2 * v + 1]);
    const float dx = ex - sx, dy = ey - sy;
    const float seg = __builtin_amdgcn_sqrtf(fmaf(dx, dx, dy * dy));

    // Clip t to the image slab (samples outside contribute exactly 0).
    float txa = (-HALF_EXT - sx) / dx, txb = (HALF_EXT - sx) / dx;
    float tya = (-HALF_EXT - sy) / dy, tyb = (HALF_EXT - sy) / dy;
    float tlo = fmaxf(fmaxf(fminf(txa, txb), fminf(tya, tyb)), 0.0f);
    float thi = fminf(fminf(fmaxf(txa, txb), fmaxf(tya, tyb)), 1.0f);
    #pragma unroll
    for (int off = 16; off; off >>= 1) {   // wave-uniform window
        tlo = fminf(tlo, __shfl_xor(tlo, off, 32));
        thi = fmaxf(thi, __shfl_xor(thi, off, 32));
    }
    int s_start = (int)floorf(tlo * (float)NSAMP - 0.5f) - 1;
    s_start = (s_start < 0) ? 0 : (s_start & ~15);
    int s_end = (int)ceilf(thi * (float)NSAMP - 0.5f) + 1;
    if (s_end > NSAMP - 1) s_end = NSAMP - 1;

    const float lane_off = (float)((l >> 4) << 3) + 0.5f; // half*8 + 0.5
    const float tscale   = 1.0f / (float)NSAMP;
    const float poff     = FUSED ? HALF_W : (HALF_W + (float)PAD);

    v16bf amat;                                  // A = ones
    #pragma unroll
    for (int j = 0; j < 16; ++j) amat[j] = (__bf16)1.0f;
    v8f c = {};

    float sbf = (float)s_start;                  // exact integer float
    for (int sb = s_start; sb <= s_end; sb += 16, sbf += 16.0f) {
        v16bf b;
        #pragma unroll
        for (int j = 0; j < 8; ++j) {
            float t  = (sbf + (lane_off + (float)j)) * tscale; // exact t grid
            float px = fmaf(t, dx, sx) + poff;
            float py = fmaf(t, dy, sy) + poff;
            float s  = FUSED ? bilin_chk(imgA, imgB, px, py)
                             : bilin_pad(imgA, px, py);
            __bf16 hi = (__bf16)s;
            b[j]     = hi;                       // K(half) = soff..soff+7 : hi
            b[8 + j] = (__bf16)(s - (float)hi);  //                        : lo
        }
        c = __builtin_amdgcn_wmma_f32_16x16x32_bf16(
                false, amat, false, b, (short)0, c, false, false);
    }

    // Lane n holds column n's sum in c[0] (rows identical). One coalesced store.
    float result = c[0] * seg * tscale;
    if (l < 16) sino[v * N_DET + d0 + m] = result;
}

// ---------------------------------------------------------------------------
extern "C" void kernel_launch(void* const* d_in, const int* in_sizes, int n_in,
                              void* d_out, int out_size, void* d_ws, size_t ws_size,
                              hipStream_t stream) {
    const float* x    = (const float*)d_in[0];
    const float* reco = (const float*)d_in[1];
    const float* src  = (const float*)d_in[2];
    const float* detc = (const float*)d_in[3];
    const float* duv  = (const float*)d_in[4];

    float* out  = (float*)d_out;
    float* sino = out;                        // 180*512
    float* relu = out + N_VIEWS * N_DET;      // 512*512

    const size_t pad_bytes = (size_t)PW * PH * sizeof(float);
    const bool staged = (d_ws != nullptr) && (ws_size >= pad_bytes);
    float* pimg = (float*)d_ws;

    dim3 pgrid((PW + 255) / 256, PH);
    dim3 grid(N_VIEWS * 4);                   // 720 blocks * 8 waves/block

    if (staged) {
        prep_padded<<<pgrid, 256, 0, stream>>>(x, reco, pimg, relu);
        proj_kernel<false><<<grid, 256, 0, stream>>>(pimg, nullptr, src, detc, duv, sino);
    } else {
        prep_simple<<<(IMG_H * IMG_W / 4) / 256, 256, 0, stream>>>(
            (const float4*)x, (const float4*)reco, (float4*)relu);
        proj_kernel<true><<<grid, 256, 0, stream>>>(x, reco, src, detc, duv, sino);
    }
}
